// GLCN_86569360818244
// MI455X (gfx1250) — matrix operand order repaired and checked by reading, not compiled, not run on hardware
//
#include <hip/hip_runtime.h>
#include <hip/hip_bf16.h>

typedef __attribute__((ext_vector_type(16))) _Float16 v16h;
typedef __attribute__((ext_vector_type(8)))  _Float16 v8h;
typedef __attribute__((ext_vector_type(8)))  float    v8f;

#define N_NODES 2048
#define FEAT    256
#define FOBS    64
#define DEMB    256
#define NEGF    (-9.0e15f)

// ---------------- link prediction: binary mask A ----------------
// A[i,j] = 1 if i==j else (sum_k |h_i[k]-h_j[k]|*a_link[k] + g[i,j] > 0)
__global__ void lp_mask_kernel(const float* __restrict__ X,
                               const float* __restrict__ a_link,
                               const float* __restrict__ g,
                               unsigned char* __restrict__ mask)
{
    __shared__ float hi[16][FOBS];
    __shared__ float hj[16][FOBS];
    __shared__ float w[FOBS];
    int tx = threadIdx.x, ty = threadIdx.y;
    int t = ty * 16 + tx;
    int i0 = blockIdx.y * 16, j0 = blockIdx.x * 16;
    for (int e = t; e < 16 * FOBS; e += 256) {
        int r = e >> 6, k = e & 63;
        hi[r][k] = X[(size_t)(i0 + r) * FEAT + k];
        hj[r][k] = X[(size_t)(j0 + r) * FEAT + k];
    }
    if (t < FOBS) w[t] = a_link[t];
    __syncthreads();
    float acc = 0.f;
    #pragma unroll 8
    for (int k = 0; k < FOBS; ++k)
        acc += fabsf(hi[ty][k] - hj[tx][k]) * w[k];
    int i = i0 + ty, j = j0 + tx;
    float z = acc + g[(size_t)i * N_NODES + j];
    mask[(size_t)i * N_NODES + j] = (i == j) ? 1 : (z > 0.f ? 1 : 0);
}

// ---------------- f32 -> f16 convert ----------------
__global__ void cvt_f32_to_f16(const float* __restrict__ in,
                               _Float16* __restrict__ out, int n)
{
    int i = blockIdx.x * 256 + threadIdx.x;
    if (i < n) out[i] = (_Float16)in[i];
}

// ---------------- transpose + convert: in f32 [R,C] -> out f16 [C,R] ----------------
__global__ void transpose_cvt_kernel(const float* __restrict__ in,
                                     _Float16* __restrict__ out, int R, int C)
{
    __shared__ float tile[16][17];
    int tx = threadIdx.x, ty = threadIdx.y;
    int r = blockIdx.y * 16 + ty, c = blockIdx.x * 16 + tx;
    tile[ty][tx] = in[(size_t)r * C + c];
    __syncthreads();
    int oc = blockIdx.x * 16 + ty;   // output row = original column
    int orr = blockIdx.y * 16 + tx;  // output col = original row
    out[(size_t)oc * R + orr] = (_Float16)tile[tx][ty];
}

// ---------------- dual GEMV: f1 = Wh @ a[0:D], f2 = Wh @ a[D:2D] ----------------
__global__ void gemv_dual_kernel(const float* __restrict__ Wh,
                                 const float* __restrict__ a,
                                 float* __restrict__ f1, float* __restrict__ f2)
{
    int row = blockIdx.x;
    int lane = threadIdx.x;  // wave32
    const float* p = Wh + (size_t)row * DEMB;
    float s1 = 0.f, s2 = 0.f;
    for (int k = lane; k < DEMB; k += 32) {
        float v = p[k];
        s1 += v * a[k];
        s2 += v * a[DEMB + k];
    }
    for (int off = 16; off > 0; off >>= 1) {
        s1 += __shfl_down(s1, off, 32);
        s2 += __shfl_down(s2, off, 32);
    }
    if (lane == 0) { f1[row] = s1; f2[row] = s2; }
}

// ---------------- masked row softmax -> att (f16, row-major) ----------------
__global__ void softmax_att_kernel(const unsigned char* __restrict__ mask,
                                   const float* __restrict__ f1,
                                   const float* __restrict__ f2,
                                   _Float16* __restrict__ att)
{
    __shared__ float red[256];
    int i = blockIdx.x;
    int t = threadIdx.x;
    float fi = f1[i];
    float ex[8];
    bool on[8];
    float vals[8];
    float mx = NEGF;
    #pragma unroll
    for (int c = 0; c < 8; ++c) {
        int j = t + 256 * c;
        bool m = mask[(size_t)i * N_NODES + j] != 0;
        float e = fi + f2[j];
        e = (e > 0.f) ? e : 0.15f * e;  // leaky_relu(0.15)
        on[c] = m; vals[c] = e;
        mx = fmaxf(mx, m ? e : NEGF);
    }
    red[t] = mx; __syncthreads();
    for (int s = 128; s > 0; s >>= 1) {
        if (t < s) red[t] = fmaxf(red[t], red[t + s]);
        __syncthreads();
    }
    mx = red[0]; __syncthreads();
    float sum = 0.f;
    #pragma unroll
    for (int c = 0; c < 8; ++c) {
        float e = on[c] ? __expf(vals[c] - mx) : 0.f;
        ex[c] = e; sum += e;
    }
    red[t] = sum; __syncthreads();
    for (int s = 128; s > 0; s >>= 1) {
        if (t < s) red[t] += red[t + s];
        __syncthreads();
    }
    float inv = 1.f / red[0];
    #pragma unroll
    for (int c = 0; c < 8; ++c) {
        int j = t + 256 * c;
        att[(size_t)i * N_NODES + j] = (_Float16)(ex[c] * inv);
    }
}

// ---------------- WMMA fragment loaders ----------------
// A-frag (16-bit A 16x32): lane m = l&15, h = l>>4; element e -> K = e+16h
//   => two contiguous 16-byte loads at (k0+16h) and (k0+16h+8) from row m.
__device__ __forceinline__ v16h load_afrag(const _Float16* __restrict__ row, int k0, int h)
{
    v8h lo = *(const v8h*)(row + k0 + 16 * h);
    v8h hi = *(const v8h*)(row + k0 + 16 * h + 8);
    v16h f;
    #pragma unroll
    for (int e = 0; e < 8; ++e) { f[e] = lo[e]; f[e + 8] = hi[e]; }
    return f;
}
// B-frag (16-bit B 32x16): lane n = l&15, h = l>>4; element e -> K = e+16h
//   => one contiguous 32-byte load from BT row n at (k0+16h).
__device__ __forceinline__ v16h load_bfrag(const _Float16* __restrict__ row, int k0, int h)
{
    return *(const v16h*)(row + k0 + 16 * h);
}

// ---------------- WMMA GEMM: C[M,N] f32 = A[M,K] f16 x BT[N,K] f16 ----------------
// One wave per 32x32 C tile: 2 A-frags x 2 B-frags -> 4 independent WMMA chains.
// Register-level operand reuse halves L2 traffic vs 16x16-per-wave; 4 independent
// accumulators hide the f16-WMMA RAW latency (tracked as TRANS -> co-executes with VALU).
__global__ __launch_bounds__(32)
void wmma_gemm_bt_kernel(const _Float16* __restrict__ A,
                         const _Float16* __restrict__ BT,
                         float* __restrict__ C,
                         int M, int N, int K, int relu)
{
    int lane = threadIdx.x;
    int mn = lane & 15;
    int h  = lane >> 4;
    const _Float16* a0p = A  + (size_t)(blockIdx.y * 32 + mn) * K;
    const _Float16* a1p = a0p + (size_t)16 * K;
    const _Float16* b0p = BT + (size_t)(blockIdx.x * 32 + mn) * K;
    const _Float16* b1p = b0p + (size_t)16 * K;

    v8f acc00 = {}, acc01 = {}, acc10 = {}, acc11 = {};
    for (int k0 = 0; k0 < K; k0 += 32) {
        // near-cache prefetch of each lane's row, 256B ahead (global_prefetch_b8)
        if (k0 + 128 < K) {
            __builtin_prefetch(a0p + k0 + 128, 0, 3);
            __builtin_prefetch(a1p + k0 + 128, 0, 3);
            __builtin_prefetch(b0p + k0 + 128, 0, 3);
            __builtin_prefetch(b1p + k0 + 128, 0, 3);
        }
        v16h af0 = load_afrag(a0p, k0, h);
        v16h af1 = load_afrag(a1p, k0, h);
        v16h bf0 = load_bfrag(b0p, k0, h);
        v16h bf1 = load_bfrag(b1p, k0, h);
        acc00 = __builtin_amdgcn_wmma_f32_16x16x32_f16(false, af0, false, bf0, (short)0, acc00, false, false);
        acc01 = __builtin_amdgcn_wmma_f32_16x16x32_f16(false, af0, false, bf1, (short)0, acc01, false, false);
        acc10 = __builtin_amdgcn_wmma_f32_16x16x32_f16(false, af1, false, bf0, (short)0, acc10, false, false);
        acc11 = __builtin_amdgcn_wmma_f32_16x16x32_f16(false, af1, false, bf1, (short)0, acc11, false, false);
    }

    // C tile: lane l -> col n = l&15 ; VGPR r -> row r + 8h  (per 16x16 subtile)
    int cm = blockIdx.y * 32;
    int cn = blockIdx.x * 32 + mn;
    #pragma unroll
    for (int r = 0; r < 8; ++r) {
        float v00 = acc00[r], v01 = acc01[r], v10 = acc10[r], v11 = acc11[r];
        if (relu) {
            v00 = fmaxf(v00, 0.f); v01 = fmaxf(v01, 0.f);
            v10 = fmaxf(v10, 0.f); v11 = fmaxf(v11, 0.f);
        }
        size_t row0 = (size_t)(cm + r + 8 * h) * N;
        size_t row1 = (size_t)(cm + 16 + r + 8 * h) * N;
        C[row0 + cn]      = v00;
        C[row0 + cn + 16] = v01;
        C[row1 + cn]      = v10;
        C[row1 + cn + 16] = v11;
    }
}

extern "C" void kernel_launch(void* const* d_in, const int* in_sizes, int n_in,
                              void* d_out, int out_size, void* d_ws, size_t ws_size,
                              hipStream_t stream)
{
    (void)in_sizes; (void)n_in; (void)out_size; (void)ws_size;
    const float* X      = (const float*)d_in[0];
    const float* a_link = (const float*)d_in[1];
    const float* W0     = (const float*)d_in[2];
    const float* W1     = (const float*)d_in[3];
    const float* a0     = (const float*)d_in[4];
    const float* a1     = (const float*)d_in[5];
    const float* g      = (const float*)d_in[6];
    float* out = (float*)d_out;

    char* ws = (char*)d_ws;
    size_t off = 0;
    auto carve = [&](size_t bytes) -> void* {
        void* p = ws + off;
        off += (bytes + 255) & ~(size_t)255;
        return p;
    };
    unsigned char* mask = (unsigned char*)carve((size_t)N_NODES * N_NODES);        // 4 MB
    _Float16* Hh  = (_Float16*)carve((size_t)N_NODES * FEAT * 2);                  // 1 MB
    _Float16* WT  = (_Float16*)carve((size_t)FEAT * DEMB * 2);                     // 128 KB
    float*    Whf = (float*)carve((size_t)N_NODES * DEMB * 4);                     // 2 MB
    _Float16* WhT = (_Float16*)carve((size_t)DEMB * N_NODES * 2);                  // 1 MB
    float*    f1  = (float*)carve((size_t)N_NODES * 4);
    float*    f2  = (float*)carve((size_t)N_NODES * 4);
    _Float16* att = (_Float16*)carve((size_t)N_NODES * N_NODES * 2);               // 8 MB
    float*    Hbuf = (float*)carve((size_t)N_NODES * DEMB * 4);                    // 2 MB

    dim3 blk16(16, 16);

    // adjacency mask (computed once)
    lp_mask_kernel<<<dim3(N_NODES / 16, N_NODES / 16), blk16, 0, stream>>>(X, a_link, g, mask);

    const float* Hcur = X;
    const float* Wl[2] = {W0, W1};
    const float* al[2] = {a0, a1};
    for (int layer = 0; layer < 2; ++layer) {
        // H -> f16
        cvt_f32_to_f16<<<(N_NODES * FEAT) / 256, 256, 0, stream>>>(Hcur, Hh, N_NODES * FEAT);
        // W[F,D] -> WT f16 [D,F]
        transpose_cvt_kernel<<<dim3(DEMB / 16, FEAT / 16), blk16, 0, stream>>>(Wl[layer], WT, FEAT, DEMB);
        // Wh = H @ W  (M=2048, N=256, K=256)
        wmma_gemm_bt_kernel<<<dim3(DEMB / 32, N_NODES / 32), 32, 0, stream>>>(
            Hh, WT, Whf, N_NODES, DEMB, FEAT, 0);
        // f1 = Wh a_lo, f2 = Wh a_hi
        gemv_dual_kernel<<<N_NODES, 32, 0, stream>>>(Whf, al[layer], f1, f2);
        // Wh[N,D] -> WhT f16 [D,N]
        transpose_cvt_kernel<<<dim3(DEMB / 16, N_NODES / 16), blk16, 0, stream>>>(Whf, WhT, N_NODES, DEMB);
        // masked softmax -> att f16
        softmax_att_kernel<<<N_NODES, 256, 0, stream>>>(mask, f1, f2, att);
        // H = relu(att @ Wh)  (M=2048, N=256, K=2048)
        float* Hout = (layer == 0) ? Hbuf : out;
        wmma_gemm_bt_kernel<<<dim3(DEMB / 32, N_NODES / 32), 32, 0, stream>>>(
            att, WhT, Hout, N_NODES, DEMB, N_NODES, 1);
        Hcur = Hbuf;
    }
}
